// TemperatureGuidedAttention_52518860095586
// MI455X (gfx1250) — matrix operand order repaired and checked by reading, not compiled, not run on hardware
//
#include <hip/hip_runtime.h>
#include <hip/hip_bf16.h>
#include <math.h>

// ---------------- CDNA5 WMMA types ----------------
typedef __bf16 v16bf __attribute__((ext_vector_type(16)));
typedef __bf16 v8bf  __attribute__((ext_vector_type(8)));
typedef float  v8f   __attribute__((ext_vector_type(8)));

__device__ __forceinline__ __bf16 f2bf(float f) {
  union { float f; unsigned u; } in; in.f = f;
  unsigned r = in.u + 0x7FFFu + ((in.u >> 16) & 1u);   // RNE
  union { unsigned short s; __bf16 b; } out;
  out.s = (unsigned short)(r >> 16);
  return out.b;
}
__device__ __forceinline__ void storev(float* p, float v)  { *p = v; }
__device__ __forceinline__ void storev(__bf16* p, float v) { *p = f2bf(v); }

#define CAT16(lo, hi) __builtin_shufflevector(lo, hi, 0,1,2,3,4,5,6,7,8,9,10,11,12,13,14,15)

#define HID 1024
#define SEQ 1024
#define NHEAD 16
#define DHEAD 64
#define MROWS 4096          // B*S

// f32 -> bf16 conversion pass (bandwidth bound, done once per tensor)
__global__ __launch_bounds__(256) void cvt_f32_bf16(
    const float* __restrict__ s, __bf16* __restrict__ d)
{
  int i = blockIdx.x * 1024 + threadIdx.x * 4;
  float4 v = *(const float4*)(s + i);
  d[i]     = f2bf(v.x);
  d[i + 1] = f2bf(v.y);
  d[i + 2] = f2bf(v.z);
  d[i + 3] = f2bf(v.w);
}

// =====================================================================
// GEMM: C[M,N] = A[M,K] * W[N,K]^T, bf16 inputs, f32 accum.
// block (32,4): 4 waves; each wave owns a 32x64 tile (8 WMMAs / k-step).
// grid (M/32, N/256). Fragments are loaded as raw b128 vectors (no cvt).
// =====================================================================
template <typename TOUT>
__global__ __launch_bounds__(128) void gemm_bf16(
    const __bf16* __restrict__ A, int lda,
    const __bf16* __restrict__ W, int ldw,
    TOUT* __restrict__ C, int ldc, int K)
{
  const int lane   = threadIdx.x;
  const int wv     = threadIdx.y;
  const int lane16 = lane & 15;
  const int laneHi = lane >> 4;
  const int row = blockIdx.x * 32;
  const int col = blockIdx.y * 256 + wv * 64;

  v8f acc[2][4] = {};
  const __bf16* arow0 = A + (size_t)(row + lane16) * lda + laneHi * 16;
  const __bf16* arow1 = arow0 + (size_t)16 * lda;

  for (int k0 = 0; k0 < K; k0 += 32) {
    __builtin_prefetch(arow0 + k0 + 256, 0, 0);   // -> global_prefetch_b8
    // A fragments: 16 contiguous elems per lane (k = 16*laneHi + s)
    v16bf af0 = CAT16(*(const v8bf*)(arow0 + k0), *(const v8bf*)(arow0 + k0 + 8));
    v16bf af1 = CAT16(*(const v8bf*)(arow1 + k0), *(const v8bf*)(arow1 + k0 + 8));
#pragma unroll
    for (int t = 0; t < 4; ++t) {
      // B fragment: 16 contiguous elems per lane; B[k][n] = W[n][k]
      v16bf bf = *(const v16bf*)(W + (size_t)(col + t * 16 + lane16) * ldw + k0 + laneHi * 16);
      acc[0][t] = __builtin_amdgcn_wmma_f32_16x16x32_bf16(false, af0, false, bf, (short)0, acc[0][t], false, false);
      acc[1][t] = __builtin_amdgcn_wmma_f32_16x16x32_bf16(false, af1, false, bf, (short)0, acc[1][t], false, false);
    }
  }
#pragma unroll
  for (int g = 0; g < 2; ++g)
#pragma unroll
    for (int t = 0; t < 4; ++t)
#pragma unroll
      for (int r = 0; r < 8; ++r)
        storev(&C[(size_t)(row + g * 16 + r + laneHi * 8) * ldc + col + t * 16 + lane16],
               acc[g][t][r]);
}

// =====================================================================
// Flash attention, bf16 Q/K/V in, bf16 ctx out, f32 softmax state.
// one wave per (b, h, 16-query tile); P staged through bf16 LDS tile.
// =====================================================================
__global__ __launch_bounds__(32) void attn_wmma(
    const __bf16* __restrict__ Qb, const __bf16* __restrict__ Kb,
    const __bf16* __restrict__ Vb, int ld,
    __bf16* __restrict__ Ob, int ldo,
    const float* __restrict__ twn, float scale)
{
  __shared__ __bf16 P[16][32];                     // 16x32 prob tile (1 KB)

  const int lane   = threadIdx.x;
  const int lane16 = lane & 15;
  const int laneHi = lane >> 4;
  const int qt = blockIdx.x, h = blockIdx.y, b = blockIdx.z;

  const __bf16* qp = Qb + (size_t)b * SEQ * ld + h * DHEAD;
  const __bf16* kp = Kb + (size_t)b * SEQ * ld + h * DHEAD;
  const __bf16* vp = Vb + (size_t)b * SEQ * ld + h * DHEAD;
  __bf16*       op = Ob + (size_t)b * SEQ * ldo + h * DHEAD;

  float fs[8];
#pragma unroll
  for (int r = 0; r < 8; ++r) {
    int qrow = qt * 16 + r + laneHi * 8;
    fs[r] = scale * (twn ? twn[b * SEQ + qrow] : 1.0f);
  }

  // Q A-fragments for dh 0..31 / 32..63, raw vector loads
  const __bf16* qr = qp + (size_t)(qt * 16 + lane16) * ld + laneHi * 16;
  v16bf aq0 = CAT16(*(const v8bf*)(qr),      *(const v8bf*)(qr + 8));
  v16bf aq1 = CAT16(*(const v8bf*)(qr + 32), *(const v8bf*)(qr + 40));

  float mrow[8], lrow[8];
  v8f o[4] = {};
#pragma unroll
  for (int r = 0; r < 8; ++r) { mrow[r] = -1e30f; lrow[r] = 0.f; }

  for (int kc = 0; kc < SEQ; kc += 32) {
    // ---- scores: two 16x16 tiles over this 32-key chunk ----
    v8f sc[2];
#pragma unroll
    for (int t = 0; t < 2; ++t) {
      const __bf16* kr = kp + (size_t)(kc + t * 16 + lane16) * ld + laneHi * 16;
      v16bf bk0 = *(const v16bf*)(kr);            // dh 0..31
      v16bf bk1 = *(const v16bf*)(kr + 32);       // dh 32..63
      v8f z = {};
      z = __builtin_amdgcn_wmma_f32_16x16x32_bf16(false, aq0, false, bk0, (short)0, z, false, false);
      z = __builtin_amdgcn_wmma_f32_16x16x32_bf16(false, aq1, false, bk1, (short)0, z, false, false);
      sc[t] = z;
    }
    // ---- online softmax update ----
    float alpha[8];
#pragma unroll
    for (int r = 0; r < 8; ++r) {
      float l0 = sc[0][r] * fs[r];
      float l1 = sc[1][r] * fs[r];
      float mx = fmaxf(l0, l1);
#pragma unroll
      for (int d = 1; d < 16; d <<= 1) mx = fmaxf(mx, __shfl_xor(mx, d, 32));
      float mnew = fmaxf(mrow[r], mx);
      float a  = __expf(mrow[r] - mnew);
      float p0 = __expf(l0 - mnew);
      float p1 = __expf(l1 - mnew);
      float s  = p0 + p1;
#pragma unroll
      for (int d = 1; d < 16; d <<= 1) s += __shfl_xor(s, d, 32);
      lrow[r] = lrow[r] * a + s;
      mrow[r] = mnew;
      alpha[r] = a;
      P[r + laneHi * 8][lane16]      = f2bf(p0);
      P[r + laneHi * 8][16 + lane16] = f2bf(p1);
    }
#pragma unroll
    for (int t = 0; t < 4; ++t)
#pragma unroll
      for (int r = 0; r < 8; ++r) o[t][r] *= alpha[r];

    __syncthreads();
    // probabilities as A-fragment: two b128 LDS reads
    const v8bf* prow = (const v8bf*)(&P[lane16][0]);
    v16bf ap = CAT16(prow[2 * laneHi], prow[2 * laneHi + 1]);
    __syncthreads();
    // ---- ctx += P * V (4 dh tiles) ----
#pragma unroll
    for (int t = 0; t < 4; ++t) {
      const __bf16* vc = vp + t * 16 + lane16;
      v16bf bv;
#pragma unroll
      for (int j = 0; j < 16; ++j)
        bv[j] = vc[(size_t)(kc + j + laneHi * 16) * ld];
      o[t] = __builtin_amdgcn_wmma_f32_16x16x32_bf16(false, ap, false, bv, (short)0, o[t], false, false);
    }
  }

#pragma unroll
  for (int t = 0; t < 4; ++t)
#pragma unroll
    for (int r = 0; r < 8; ++r) {
      int qrow = qt * 16 + r + laneHi * 8;
      op[(size_t)qrow * ldo + t * 16 + lane16] = f2bf(o[t][r] / lrow[r]);
    }
}

// ---------------- row-kernel helpers ----------------
__device__ __forceinline__ float block_reduce_sum(float v, float* red, int nwaves) {
  int tid = threadIdx.x;
#pragma unroll
  for (int d = 1; d < 32; d <<= 1) v += __shfl_xor(v, d, 32);
  if ((tid & 31) == 0) red[tid >> 5] = v;
  __syncthreads();
  if (tid == 0) {
    float s = 0.f;
    for (int w = 0; w < nwaves; ++w) s += red[w];
    red[0] = s;
  }
  __syncthreads();
  float out = red[0];
  __syncthreads();
  return out;
}
__device__ __forceinline__ float gelu_exact(float x) {
  return 0.5f * x * (1.f + erff(x * 0.70710678118654752f));
}
__device__ __forceinline__ float sigmoidf(float x) {
  return 1.f / (1.f + __expf(-x));
}

// =====================================================================
// ttm tail: temp16 = sigmoid(LN16(ttm_out . wt^T)); T0 = 1+0.1*(temp-.5);
// T_base = sigmoid(gelu(LN1024(T0 . thp_w^T + b))); temps[0] -> tail.
// =====================================================================
__global__ __launch_bounds__(256) void ttm_small_kernel(
    const float* __restrict__ ttm_out, const float* __restrict__ wt,
    const float* __restrict__ g16, const float* __restrict__ b16,
    const float* __restrict__ thp_w, const float* __restrict__ thp_b,
    const float* __restrict__ g1024, const float* __restrict__ b1024,
    float* __restrict__ T_base, float* __restrict__ tail)
{
  __shared__ float t16[16];
  __shared__ float T0[16];
  __shared__ float red[8];
  const int row = blockIdx.x, tid = threadIdx.x;
  const float* xr = ttm_out + (size_t)row * HID;

  const int g = tid >> 4, lg = tid & 15;
  {
    const float* wr = wt + (size_t)g * HID;
    float s = 0.f;
    for (int i = lg; i < HID; i += 16) s += xr[i] * wr[i];
#pragma unroll
    for (int d = 1; d < 16; d <<= 1) s += __shfl_xor(s, d, 32);
    if (lg == 0) t16[g] = s;
  }
  __syncthreads();
  if (tid < 16) {
    float m = 0.f, v = 0.f;
    for (int i = 0; i < 16; ++i) m += t16[i];
    m *= (1.f / 16.f);
    for (int i = 0; i < 16; ++i) { float d = t16[i] - m; v += d * d; }
    v *= (1.f / 16.f);
    float n = (t16[tid] - m) * rsqrtf(v + 1e-5f) * g16[tid] + b16[tid];
    T0[tid] = 1.0f + 0.1f * (sigmoidf(n) - 0.5f);
  }
  __syncthreads();

  float y[4], s = 0.f, s2 = 0.f;
#pragma unroll
  for (int p = 0; p < 4; ++p) {
    int idx = tid + 256 * p;
    const float* w = thp_w + (size_t)idx * 16;
    float acc = thp_b[idx];
#pragma unroll
    for (int n = 0; n < 16; ++n) acc += T0[n] * w[n];
    y[p] = acc; s += acc; s2 += acc * acc;
  }
  s  = block_reduce_sum(s,  red, 8);
  s2 = block_reduce_sum(s2, red, 8);
  float mean = s * (1.f / 1024.f);
  float var  = s2 * (1.f / 1024.f) - mean * mean;
  float inv  = rsqrtf(var + 1e-5f);
  float* tb = T_base + (size_t)row * HID;
#pragma unroll
  for (int p = 0; p < 4; ++p) {
    int idx = tid + 256 * p;
    float n = (y[p] - mean) * inv * g1024[idx] + b1024[idx];
    float out = sigmoidf(gelu_exact(n));
    tb[idx] = out;
    if (idx == 0) {
      int bb = row >> 10, sidx = row & 1023;
      tail[(size_t)bb * 3 * SEQ + sidx] = out;       // temps[0]
    }
  }
}

// comb_bf[row] = [ xbf[row] | bf16(Tvec[row]) or broadcast bf16(Tscal[row]) ]
__global__ __launch_bounds__(256) void concat_kernel(
    const __bf16* __restrict__ xbf, const float* __restrict__ Tvec,
    const float* __restrict__ Tscal, __bf16* __restrict__ comb)
{
  const int row = blockIdx.x, tid = threadIdx.x;
  const __bf16* xr = xbf + (size_t)row * HID;
  __bf16* cr = comb + (size_t)row * 2 * HID;
  __bf16 tval = Tvec ? f2bf(0.f) : f2bf(Tscal[row]);
#pragma unroll
  for (int p = 0; p < 4; ++p) {
    int i = tid + 256 * p;
    cr[i] = xr[i];
    cr[HID + i] = Tvec ? f2bf(Tvec[(size_t)row * HID + i]) : tval;
  }
}

// h = gelu(LN1024(h + bias)) in place
__global__ __launch_bounds__(256) void ln_gelu_kernel(
    float* __restrict__ hbuf, const float* __restrict__ bias,
    const float* __restrict__ g, const float* __restrict__ b)
{
  __shared__ float red[8];
  const int row = blockIdx.x, tid = threadIdx.x;
  float* hr = hbuf + (size_t)row * HID;
  float y[4], s = 0.f, s2 = 0.f;
#pragma unroll
  for (int p = 0; p < 4; ++p) {
    int i = tid + 256 * p;
    float v = hr[i] + bias[i];
    y[p] = v; s += v; s2 += v * v;
  }
  s  = block_reduce_sum(s,  red, 8);
  s2 = block_reduce_sum(s2, red, 8);
  float mean = s * (1.f / 1024.f);
  float var  = s2 * (1.f / 1024.f) - mean * mean;
  float inv  = rsqrtf(var + 1e-5f);
#pragma unroll
  for (int p = 0; p < 4; ++p) {
    int i = tid + 256 * p;
    float n = (y[p] - mean) * inv * g[i] + b[i];
    hr[i] = gelu_exact(n);
  }
}

// Ti = sigmoid(sigmoid(h . w + b2))
__global__ __launch_bounds__(256) void gemv_sig_kernel(
    const float* __restrict__ hbuf, const float* __restrict__ w,
    const float* __restrict__ b2, float* __restrict__ Ti,
    float* __restrict__ tail, int iter)
{
  __shared__ float red[8];
  const int row = blockIdx.x, tid = threadIdx.x;
  const float* hr = hbuf + (size_t)row * HID;
  float s = 0.f;
#pragma unroll
  for (int p = 0; p < 4; ++p) { int i = tid + 256 * p; s += hr[i] * w[i]; }
  s = block_reduce_sum(s, red, 8);
  if (tid == 0) {
    float v = sigmoidf(sigmoidf(s + b2[0]));
    Ti[row] = v;
    int bb = row >> 10, sidx = row & 1023;
    tail[(size_t)bb * 3 * SEQ + (size_t)iter * SEQ + sidx] = v;
  }
}

// tw = mean of 3 temps; global mean + unbiased std over Nt=B*NH*S*S; twn
__global__ __launch_bounds__(1024) void stats_kernel(
    const float* __restrict__ tail, float* __restrict__ twn)
{
  __shared__ float red[32];
  __shared__ float mv[2];
  const int tid = threadIdx.x;
  float tw[4], s = 0.f, s2 = 0.f;
#pragma unroll
  for (int p = 0; p < 4; ++p) {
    int row = tid + 1024 * p;
    int bb = row >> 10, sidx = row & 1023;
    const float* tb = tail + (size_t)bb * 3 * SEQ;
    float v = (tb[sidx] + tb[SEQ + sidx] + tb[2 * SEQ + sidx]) * (1.f / 3.f);
    tw[p] = v; s += v; s2 += v * v;
  }
#pragma unroll
  for (int d = 1; d < 32; d <<= 1) { s += __shfl_xor(s, d, 32); s2 += __shfl_xor(s2, d, 32); }
  if ((tid & 31) == 0) red[tid >> 5] = s;
  __syncthreads();
  if (tid == 0) { float t = 0.f; for (int w = 0; w < 32; ++w) t += red[w]; mv[0] = t; }
  __syncthreads();
  if ((tid & 31) == 0) red[tid >> 5] = s2;
  __syncthreads();
  if (tid == 0) { float t = 0.f; for (int w = 0; w < 32; ++w) t += red[w]; mv[1] = t; }
  __syncthreads();
  float mu  = mv[0] * (1.f / 4096.f);
  float var = mv[1] * (1.f / 4096.f) - mu * mu;
  const double Nt = 4.0 * 16.0 * 1024.0 * 1024.0;
  float stdu = sqrtf((float)((double)var * (Nt / (Nt - 1.0))));
  float denom = stdu + 1.1920928955078125e-07f;        // float eps
#pragma unroll
  for (int p = 0; p < 4; ++p) {
    int row = tid + 1024 * p;
    twn[row] = 1.f + (tw[p] - mu) / denom;
  }
}

// =====================================================================
extern "C" void kernel_launch(void* const* d_in, const int* in_sizes, int n_in,
                              void* d_out, int out_size, void* d_ws, size_t ws_size,
                              hipStream_t stream)
{
  const float* x        = (const float*)d_in[0];
  const float* wq       = (const float*)d_in[1];
  const float* wk       = (const float*)d_in[2];
  const float* wv       = (const float*)d_in[3];
  const float* wo       = (const float*)d_in[4];
  const float* mha_in_w = (const float*)d_in[5];
  const float* mha_out_w= (const float*)d_in[6];
  const float* wt       = (const float*)d_in[7];
  const float* ttm_ln_g = (const float*)d_in[8];
  const float* ttm_ln_b = (const float*)d_in[9];
  const float* thp_w    = (const float*)d_in[10];
  const float* thp_b    = (const float*)d_in[11];
  const float* thp_ln_g = (const float*)d_in[12];
  const float* thp_ln_b = (const float*)d_in[13];
  const float* tn_w1    = (const float*)d_in[14];
  const float* tn_b1    = (const float*)d_in[15];
  const float* tn_ln_g  = (const float*)d_in[16];
  const float* tn_ln_b  = (const float*)d_in[17];
  const float* tn_w2    = (const float*)d_in[18];
  const float* tn_b2    = (const float*)d_in[19];

  float* out = (float*)d_out;

  // ---- scratch carve ----
  __bf16* xbf   = (__bf16*)d_ws;                   // 4096*1024
  __bf16* winb  = xbf   + (size_t)MROWS * HID;     // 3072*1024
  __bf16* woutb = winb  + (size_t)3 * HID * HID;   // 1024*1024
  __bf16* w1b   = woutb + (size_t)HID * HID;       // 1024*2048
  __bf16* wqb   = w1b   + (size_t)2 * HID * HID;
  __bf16* wkb   = wqb   + (size_t)HID * HID;
  __bf16* wvb   = wkb   + (size_t)HID * HID;
  __bf16* wob   = wvb   + (size_t)HID * HID;
  __bf16* qkvb  = wob   + (size_t)HID * HID;       // 4096*3072 (later q/k/v)
  __bf16* mctxb = qkvb  + (size_t)MROWS * 3 * HID; // 4096*1024 (later ctx)
  __bf16* combb = mctxb + (size_t)MROWS * HID;     // 4096*2048
  float*  buf2  = (float*)(combb + (size_t)MROWS * 2 * HID); // ttm_out / hmid
  float*  tbase = buf2  + (size_t)MROWS * HID;
  float*  twn   = tbase + (size_t)MROWS * HID;
  float*  tibuf = twn   + MROWS;
  __bf16* qb = qkvb;
  __bf16* kb = qkvb + (size_t)MROWS * HID;
  __bf16* vb = qkvb + (size_t)2 * MROWS * HID;
  float*  tail = out + (size_t)MROWS * HID;        // scale_temps [B,3,S,1]

  const float scale = 0.125f;                      // DH^-0.5
  dim3 gblk(32, 4, 1);
  dim3 agrid(SEQ / 16, NHEAD, 4);

  // ---- one-time f32 -> bf16 passes ----
  cvt_f32_bf16<<<MROWS * HID / 1024, 256, 0, stream>>>(x, xbf);
  cvt_f32_bf16<<<3 * HID * HID / 1024, 256, 0, stream>>>(mha_in_w, winb);
  cvt_f32_bf16<<<HID * HID / 1024, 256, 0, stream>>>(mha_out_w, woutb);
  cvt_f32_bf16<<<2 * HID * HID / 1024, 256, 0, stream>>>(tn_w1, w1b);
  cvt_f32_bf16<<<HID * HID / 1024, 256, 0, stream>>>(wq, wqb);
  cvt_f32_bf16<<<HID * HID / 1024, 256, 0, stream>>>(wk, wkb);
  cvt_f32_bf16<<<HID * HID / 1024, 256, 0, stream>>>(wv, wvb);
  cvt_f32_bf16<<<HID * HID / 1024, 256, 0, stream>>>(wo, wob);

  // ---- compute_ttm ----
  gemm_bf16<__bf16><<<dim3(MROWS/32, 3*HID/256, 1), gblk, 0, stream>>>(
      xbf, HID, winb, HID, qkvb, 3*HID, HID);
  attn_wmma<<<agrid, dim3(32,1,1), 0, stream>>>(
      qkvb, qkvb + HID, qkvb + 2*HID, 3*HID, mctxb, HID, nullptr, scale);
  gemm_bf16<float><<<dim3(MROWS/32, HID/256, 1), gblk, 0, stream>>>(
      mctxb, HID, woutb, HID, buf2, HID, HID);
  ttm_small_kernel<<<MROWS, 256, 0, stream>>>(
      buf2, wt, ttm_ln_g, ttm_ln_b, thp_w, thp_b, thp_ln_g, thp_ln_b, tbase, tail);

  // ---- compute_gsot: two temp-net iterations ----
  concat_kernel<<<MROWS, 256, 0, stream>>>(xbf, tbase, nullptr, combb);
  gemm_bf16<float><<<dim3(MROWS/32, HID/256, 1), gblk, 0, stream>>>(
      combb, 2*HID, w1b, 2*HID, buf2, HID, 2*HID);
  ln_gelu_kernel<<<MROWS, 256, 0, stream>>>(buf2, tn_b1, tn_ln_g, tn_ln_b);
  gemv_sig_kernel<<<MROWS, 256, 0, stream>>>(buf2, tn_w2, tn_b2, tibuf, tail, 1);

  concat_kernel<<<MROWS, 256, 0, stream>>>(xbf, nullptr, tibuf, combb);
  gemm_bf16<float><<<dim3(MROWS/32, HID/256, 1), gblk, 0, stream>>>(
      combb, 2*HID, w1b, 2*HID, buf2, HID, 2*HID);
  ln_gelu_kernel<<<MROWS, 256, 0, stream>>>(buf2, tn_b1, tn_ln_g, tn_ln_b);
  gemv_sig_kernel<<<MROWS, 256, 0, stream>>>(buf2, tn_w2, tn_b2, tibuf, tail, 2);

  // ---- temperature weight normalization ----
  stats_kernel<<<1, 1024, 0, stream>>>(tail, twn);

  // ---- main attention ----
  gemm_bf16<__bf16><<<dim3(MROWS/32, HID/256, 1), gblk, 0, stream>>>(
      xbf, HID, wqb, HID, qb, HID, HID);
  gemm_bf16<__bf16><<<dim3(MROWS/32, HID/256, 1), gblk, 0, stream>>>(
      xbf, HID, wkb, HID, kb, HID, HID);
  gemm_bf16<__bf16><<<dim3(MROWS/32, HID/256, 1), gblk, 0, stream>>>(
      xbf, HID, wvb, HID, vb, HID, HID);
  attn_wmma<<<agrid, dim3(32,1,1), 0, stream>>>(
      qb, kb, vb, HID, mctxb, HID, twn, scale);
  gemm_bf16<float><<<dim3(MROWS/32, HID/256, 1), gblk, 0, stream>>>(
      mctxb, HID, wob, HID, out, HID, HID);
}
